// CausalSelfAttention_5068061409451
// MI455X (gfx1250) — compile-verified, matrix operand order checked by reading, not run on hardware
//
#include <hip/hip_runtime.h>

typedef __attribute__((ext_vector_type(16))) _Float16 v16h;
typedef __attribute__((ext_vector_type(8)))  _Float16 v8h;
typedef __attribute__((ext_vector_type(4)))  _Float16 v4h;
typedef __attribute__((ext_vector_type(8)))  float    v8f;

#define T_SEQ   4096
#define C_EMB   1024
#define NHEAD   16
#define HDIM    64

// Load one lane's v16h operand for a 16x32 f16 WMMA tile.
// p points at the start of this lane's 32-element K-chunk (row of A, or row of B^T).
// Per CDNA5 ISA layout: halves 0..7 hold K = h*8 .. h*8+7, halves 8..15 hold K = 16+h*8..+7.
__device__ __forceinline__ v16h load_row32(const _Float16* p, int h) {
  v8h lo = *(const v8h*)(p + h * 8);
  v8h hi = *(const v8h*)(p + 16 + h * 8);
  return __builtin_shufflevector(lo, hi, 0,1,2,3,4,5,6,7,8,9,10,11,12,13,14,15);
}

__device__ __forceinline__ v8f wmma16(v16h a, v16h b, v8f c) {
  return __builtin_amdgcn_wmma_f32_16x16x32_f16(false, a, false, b, (short)0, c, false, false);
}

// ---------------- conversion kernels ----------------

__global__ void __launch_bounds__(256) cvt_f32_f16(const float* __restrict__ in,
                                                   _Float16* __restrict__ out, int n4) {
  int i = blockIdx.x * blockDim.x + threadIdx.x;
  if (i < n4) {
    float4 v = ((const float4*)in)[i];
    v4h o;
    o[0] = (_Float16)v.x; o[1] = (_Float16)v.y; o[2] = (_Float16)v.z; o[3] = (_Float16)v.w;
    ((v4h*)out)[i] = o;
  }
}

// out[n*K + k] = (f16) in[k*N + n]   (K,N multiples of 32)
__global__ void __launch_bounds__(1024) transpose_cvt(const float* __restrict__ in,
                                                      _Float16* __restrict__ out,
                                                      int K, int N) {
  __shared__ float tile[32][33];
  int k0 = blockIdx.x * 32, n0 = blockIdx.y * 32;
  int tx = threadIdx.x, ty = threadIdx.y;
  tile[ty][tx] = in[(size_t)(k0 + ty) * N + (n0 + tx)];
  __syncthreads();
  out[(size_t)(n0 + ty) * K + (k0 + tx)] = (_Float16)tile[tx][ty];
}

// ---------------- shared GEMM inner body: 32x64 wave tile, double-buffered ----------------
// ap: A row block base (row l), rows l and l+16 used. bp: B^T row block base (row l).
// All operand addresses are bp/ap + constant offsets -> immediate-offset b128 loads.
__device__ __forceinline__ void gemm_body_32x64(const _Float16* ap, const _Float16* bp,
                                                int h, v8f (&acc)[2][4]) {
  const int RS = 16 * C_EMB;  // 16-row stride in elements (32 KB in bytes)
  v16h a0 = load_row32(ap, h);
  v16h a1 = load_row32(ap + RS, h);
  v16h b[4];
  #pragma unroll
  for (int nt = 0; nt < 4; ++nt) b[nt] = load_row32(bp + nt * RS, h);

  for (int k0 = 32; k0 < C_EMB; k0 += 32) {
    // prefetch next K-chunk operands into fresh registers (overlaps with WMMAs below)
    v16h a0n = load_row32(ap + k0, h);
    v16h a1n = load_row32(ap + RS + k0, h);
    v16h bn[4];
    #pragma unroll
    for (int nt = 0; nt < 4; ++nt) bn[nt] = load_row32(bp + nt * RS + k0, h);

    #pragma unroll
    for (int nt = 0; nt < 4; ++nt) {
      acc[0][nt] = wmma16(a0, b[nt], acc[0][nt]);
      acc[1][nt] = wmma16(a1, b[nt], acc[1][nt]);
    }
    a0 = a0n; a1 = a1n;
    #pragma unroll
    for (int nt = 0; nt < 4; ++nt) b[nt] = bn[nt];
  }
  // last chunk (no prefetch)
  #pragma unroll
  for (int nt = 0; nt < 4; ++nt) {
    acc[0][nt] = wmma16(a0, b[nt], acc[0][nt]);
    acc[1][nt] = wmma16(a1, b[nt], acc[1][nt]);
  }
}

// ---------------- QKV GEMM: [4096,1024] x [1024,3072] ----------------
// Wave tile 32x64; block = 4 waves (1 wave/SIMD -> max VGPR budget) = 128 rows.
__global__ void __launch_bounds__(128) qkv_gemm(const _Float16* __restrict__ x16,
                                                const _Float16* __restrict__ wT,
                                                const float* __restrict__ bias,
                                                _Float16* __restrict__ q16,
                                                _Float16* __restrict__ k16,
                                                _Float16* __restrict__ vT16) {
  const int lane = threadIdx.x & 31, l = lane & 15, h = lane >> 4;
  const int wave = threadIdx.x >> 5;
  const int rowBase = blockIdx.x * 128 + wave * 32;
  const int colBase = blockIdx.y * 64;

  v8f acc[2][4] = {};
  gemm_body_32x64(x16 + (size_t)(rowBase + l) * C_EMB,
                  wT + (size_t)(colBase + l) * C_EMB, h, acc);

  const int seg = colBase >> 10;  // 0=Q, 1=K, 2=V (uniform per block)
  #pragma unroll
  for (int nt = 0; nt < 4; ++nt) {
    int col = colBase + nt * 16 + l;
    float bv = bias[col];
    #pragma unroll
    for (int mt = 0; mt < 2; ++mt) {
      #pragma unroll
      for (int r = 0; r < 8; ++r) {
        int t = rowBase + mt * 16 + r + h * 8;
        float v = acc[mt][nt][r] + bv;
        if (seg == 0) {
          int hd = col >> 6, dd = col & 63;
          q16[((size_t)hd * T_SEQ + t) * HDIM + dd] = (_Float16)(v * 0.125f);  // 1/sqrt(64)
        } else if (seg == 1) {
          int cc = col - 1024, hd = cc >> 6, dd = cc & 63;
          k16[((size_t)hd * T_SEQ + t) * HDIM + dd] = (_Float16)v;
        } else {
          int cc = col - 2048, hd = cc >> 6, dd = cc & 63;
          vT16[((size_t)hd * HDIM + dd) * T_SEQ + t] = (_Float16)v;
        }
      }
    }
  }
}

// ---------------- causal flash attention ----------------
// grid(64, 16), block 128 (4 waves). Each wave owns 16 query rows, fully independent.
__global__ void __launch_bounds__(128) attn(const _Float16* __restrict__ q16,
                                            const _Float16* __restrict__ k16,
                                            const _Float16* __restrict__ vT16,
                                            _Float16* __restrict__ o16) {
  __shared__ alignas(16) _Float16 pb[4][16][64];  // per-wave P staging (C-layout -> A-layout)
  const int lane = threadIdx.x & 31, l = lane & 15, h = lane >> 4;
  const int wave = threadIdx.x >> 5;
  const int head = blockIdx.y;
  const int base_q = blockIdx.x * 64 + wave * 16;

  const _Float16* Q  = q16  + (size_t)head * T_SEQ * HDIM;
  const _Float16* K  = k16  + (size_t)head * T_SEQ * HDIM;
  const _Float16* Vt = vT16 + (size_t)head * HDIM * T_SEQ;

  const _Float16* qrow = Q + (size_t)(base_q + l) * HDIM;
  v16h qa0 = load_row32(qrow, h);
  v16h qa1 = load_row32(qrow + 32, h);

  v8f o[4] = {};
  float m[8], lsum[8];
  #pragma unroll
  for (int r = 0; r < 8; ++r) { m[r] = -3.0e38f; lsum[r] = 0.0f; }

  const int jt_max = (base_q + 15) >> 6;
  for (int jt = 0; jt <= jt_max; ++jt) {
    // ---- load all 4 K tiles up front (16 b128 in flight) ----
    v16h kb0[4], kb1[4];
    #pragma unroll
    for (int nt = 0; nt < 4; ++nt) {
      const _Float16* krow = K + (size_t)(jt * 64 + nt * 16 + l) * HDIM;
      kb0[nt] = load_row32(krow, h);
      kb1[nt] = load_row32(krow + 32, h);
    }
    // prefetch next j-tile's K rows (each lane pulls one row's cacheline)
    if (jt < jt_max)
      __builtin_prefetch(K + (size_t)(jt * 64 + 64 + lane * 2) * HDIM, 0, 3);

    // ---- S = Q * K^T ----
    v8f s[4];
    #pragma unroll
    for (int nt = 0; nt < 4; ++nt) {
      v8f a = {};
      a = wmma16(qa0, kb0[nt], a);
      a = wmma16(qa1, kb1[nt], a);
      s[nt] = a;
    }

    // ---- causal mask (only near diagonal) ----
    if (jt * 64 + 63 > base_q) {
      #pragma unroll
      for (int nt = 0; nt < 4; ++nt) {
        int col = jt * 64 + nt * 16 + l;
        #pragma unroll
        for (int r = 0; r < 8; ++r) {
          int row = base_q + r + h * 8;
          if (col > row) s[nt][r] = -1.0e30f;
        }
      }
    }

    // ---- online softmax (row r lives in this lane's 16-lane half-group) ----
    float mnew[8], alpha[8], rs[8];
    #pragma unroll
    for (int r = 0; r < 8; ++r) {
      float v = fmaxf(fmaxf(s[0][r], s[1][r]), fmaxf(s[2][r], s[3][r]));
      v = fmaxf(v, __shfl_xor(v, 1));
      v = fmaxf(v, __shfl_xor(v, 2));
      v = fmaxf(v, __shfl_xor(v, 4));
      v = fmaxf(v, __shfl_xor(v, 8));
      mnew[r]  = fmaxf(m[r], v);
      alpha[r] = __expf(m[r] - mnew[r]);
      m[r]     = mnew[r];
      rs[r]    = 0.0f;
    }
    #pragma unroll
    for (int nt = 0; nt < 4; ++nt)
      #pragma unroll
      for (int r = 0; r < 8; ++r) {
        float p = __expf(s[nt][r] - mnew[r]);
        s[nt][r] = p;
        rs[r] += p;
      }
    #pragma unroll
    for (int r = 0; r < 8; ++r) {
      float v = rs[r];
      v += __shfl_xor(v, 1); v += __shfl_xor(v, 2);
      v += __shfl_xor(v, 4); v += __shfl_xor(v, 8);
      lsum[r] = lsum[r] * alpha[r] + v;
    }
    #pragma unroll
    for (int nt = 0; nt < 4; ++nt)
      #pragma unroll
      for (int r = 0; r < 8; ++r)
        o[nt][r] *= alpha[r];

    // ---- P: C-layout regs -> LDS (K tiles are dead now) ----
    #pragma unroll
    for (int nt = 0; nt < 4; ++nt)
      #pragma unroll
      for (int r = 0; r < 8; ++r)
        pb[wave][r + h * 8][nt * 16 + l] = (_Float16)s[nt][r];

    // ---- V loads issue here; their latency is hidden by the LDS round-trip ----
    v16h vb0[4], vb1[4];
    #pragma unroll
    for (int nt = 0; nt < 4; ++nt) {
      const _Float16* vrow = Vt + (size_t)(nt * 16 + l) * T_SEQ + jt * 64;
      vb0[nt] = load_row32(vrow, h);
      vb1[nt] = load_row32(vrow + 32, h);
    }

    asm volatile("s_wait_dscnt 0" ::: "memory");
    const _Float16* prow = &pb[wave][l][0];
    v16h pa0 = load_row32(prow, h);
    v16h pa1 = load_row32(prow + 32, h);

    // ---- O += P * V ----
    #pragma unroll
    for (int nt = 0; nt < 4; ++nt) {
      o[nt] = wmma16(pa0, vb0[nt], o[nt]);
      o[nt] = wmma16(pa1, vb1[nt], o[nt]);
    }
  }

  float inv[8];
  #pragma unroll
  for (int r = 0; r < 8; ++r) inv[r] = 1.0f / lsum[r];
  #pragma unroll
  for (int nt = 0; nt < 4; ++nt)
    #pragma unroll
    for (int r = 0; r < 8; ++r) {
      int t = base_q + r + h * 8;
      o16[(size_t)t * C_EMB + head * HDIM + nt * 16 + l] = (_Float16)(o[nt][r] * inv[r]);
    }
}

// ---------------- output projection: [4096,1024] x [1024,1024] + bias -> f32 ----------------
__global__ void __launch_bounds__(128) proj_gemm(const _Float16* __restrict__ a16,
                                                 const _Float16* __restrict__ wT,
                                                 const float* __restrict__ bias,
                                                 float* __restrict__ out) {
  const int lane = threadIdx.x & 31, l = lane & 15, h = lane >> 4;
  const int wave = threadIdx.x >> 5;
  const int rowBase = blockIdx.x * 128 + wave * 32;
  const int colBase = blockIdx.y * 64;

  v8f acc[2][4] = {};
  gemm_body_32x64(a16 + (size_t)(rowBase + l) * C_EMB,
                  wT + (size_t)(colBase + l) * C_EMB, h, acc);

  #pragma unroll
  for (int nt = 0; nt < 4; ++nt) {
    int col = colBase + nt * 16 + l;
    float bv = bias[col];
    #pragma unroll
    for (int mt = 0; mt < 2; ++mt)
      #pragma unroll
      for (int r = 0; r < 8; ++r) {
        int t = rowBase + mt * 16 + r + h * 8;
        out[(size_t)t * C_EMB + col] = acc[mt][nt][r] + bv;
      }
  }
}

// ---------------- launch ----------------
extern "C" void kernel_launch(void* const* d_in, const int* in_sizes, int n_in,
                              void* d_out, int out_size, void* d_ws, size_t ws_size,
                              hipStream_t stream) {
  const float* x      = (const float*)d_in[0];
  const float* w_qkv  = (const float*)d_in[1];
  const float* b_qkv  = (const float*)d_in[2];
  const float* w_proj = (const float*)d_in[3];
  const float* b_proj = (const float*)d_in[4];
  float* out = (float*)d_out;

  char* ws = (char*)d_ws;
  _Float16* x16    = (_Float16*)(ws + (size_t)0);           // 8 MB
  _Float16* wqkvT  = (_Float16*)(ws + ((size_t)8  << 20));  // 6 MB
  _Float16* wprojT = (_Float16*)(ws + ((size_t)14 << 20));  // 2 MB
  _Float16* q16    = (_Float16*)(ws + ((size_t)16 << 20));  // 8 MB
  _Float16* k16    = (_Float16*)(ws + ((size_t)24 << 20));  // 8 MB
  _Float16* vT16   = (_Float16*)(ws + ((size_t)32 << 20));  // 8 MB
  _Float16* att16  = (_Float16*)(ws + ((size_t)40 << 20));  // 8 MB

  int n4 = T_SEQ * C_EMB / 4;
  cvt_f32_f16<<<dim3((n4 + 255) / 256), dim3(256), 0, stream>>>(x, x16, n4);
  transpose_cvt<<<dim3(C_EMB / 32, 3 * C_EMB / 32), dim3(32, 32), 0, stream>>>(w_qkv, wqkvT, C_EMB, 3 * C_EMB);
  transpose_cvt<<<dim3(C_EMB / 32, C_EMB / 32), dim3(32, 32), 0, stream>>>(w_proj, wprojT, C_EMB, C_EMB);

  qkv_gemm<<<dim3(T_SEQ / 128, 3 * C_EMB / 64), dim3(128), 0, stream>>>(x16, wqkvT, b_qkv, q16, k16, vT16);
  attn<<<dim3(T_SEQ / 64, NHEAD), dim3(128), 0, stream>>>(q16, k16, vT16, att16);
  proj_gemm<<<dim3(T_SEQ / 128, C_EMB / 64), dim3(128), 0, stream>>>(att16, wprojT, b_proj, out);
}
